// AttentionChunkEncoder_20375324852225
// MI455X (gfx1250) — matrix operand order repaired and checked by reading, not compiled, not run on hardware
//
#include <hip/hip_runtime.h>
#include <hip/hip_bf16.h>
#include <stdint.h>

// ---------------------------------------------------------------------------
// AttentionChunkEncoder for MI455X (gfx1250, wave32, WMMA 16x16x32 f16 + TDM)
//   B=8192, W=512, RSS=12, NLED=16, LED_F=8, HID=64
// One workgroup (256 thr = 8 waves) per batch row.
//   - rss tile staged to LDS by the Tensor Data Mover (tensor_load_to_lds)
//   - conv1 / conv2 / query projection on WMMA, attention tail on VALU
// ---------------------------------------------------------------------------

typedef _Float16 v16h __attribute__((ext_vector_type(16)));
typedef _Float16 h8   __attribute__((ext_vector_type(8)));
typedef float    v8f  __attribute__((ext_vector_type(8)));
typedef unsigned int u32x4 __attribute__((ext_vector_type(4)));
typedef int          i32x8 __attribute__((ext_vector_type(8)));
typedef int          i32x4 __attribute__((ext_vector_type(4)));

#define W_LEN   512
#define RSS_C   12
#define NLED    16
#define LED_FN  8
#define HID     64

// ---- d_ws layout (bytes) ----
#define WS_W1T   0u        // [3][32][32] f16  (Bt layout, K padded 12->32)
#define WS_W2T   6144u     // [3][64][32] f16
#define WS_QWH   18432u    // [64][512]   f16  (q_w first 512 cols)
#define WS_KEYS  83968u    // [16][64]    f32  (precomputed keys)

// ---- dynamic LDS layout (bytes) ----
#define L_XPAD   0u        // [514][32] f16 = 32896 (x, pos-major, padded)
#define L_XT     32896u    // [16][512] f16 = 16384 (x transposed, rows 12..15 zero)
#define L_H1     49280u    // [514][32] f16 = 32896 (conv1 out; first 24576 B alias
                           //                        the TDM f32 staging buffer)
#define L_TOTAL  82176u

__device__ __forceinline__ v16h load_frag16(const _Float16* p0, const _Float16* p1) {
  h8 lo = *(const h8*)p0;
  h8 hi = *(const h8*)p1;
  return __builtin_shufflevector(lo, hi, 0,1,2,3,4,5,6,7,8,9,10,11,12,13,14,15);
}

// ---------------------------------------------------------------------------
// Setup: convert weights to f16 Bt layouts in d_ws; precompute keys[16][64].
// ---------------------------------------------------------------------------
__global__ __launch_bounds__(256)
void enc_setup(const float* __restrict__ c1w,   // [32][12][3]
               const float* __restrict__ c2w,   // [64][32][3]
               const float* __restrict__ qw,    // [64][515]
               const float* __restrict__ led_feats, // [16][8]
               const float* __restrict__ led_pos,   // [16][3]
               const float* __restrict__ kw,    // [64][11]
               const float* __restrict__ kb,    // [64]
               _Float16* __restrict__ w1t,
               _Float16* __restrict__ w2t,
               _Float16* __restrict__ qwh,
               float* __restrict__ keys)
{
  int i = blockIdx.x * 256 + threadIdx.x;
  if (i < 3072) {                       // w1t[t][n][k] = c1w[n][k][t] (k<12)
    int t = i >> 10, n = (i >> 5) & 31, k = i & 31;
    w1t[i] = (k < RSS_C) ? (_Float16)c1w[(n * RSS_C + k) * 3 + t] : (_Float16)0.f;
  } else if (i < 9216) {                // w2t[t][n][k] = c2w[n][k][t]
    int j = i - 3072;
    int t = j >> 11, n = (j >> 5) & 63, k = j & 31;
    w2t[j] = (_Float16)c2w[(n * 32 + k) * 3 + t];
  } else if (i < 41984) {               // qwh[o][k] = qw[o][k], k<512
    int j = i - 9216;
    int o = j >> 9, k = j & 511;
    qwh[j] = (_Float16)qw[o * 515 + k];
  } else if (i < 43008) {               // keys[n][o]
    int j = i - 41984;
    int n = j >> 6, o = j & 63;
    float s = kb[o];
#pragma unroll
    for (int d = 0; d < LED_FN; ++d) s += led_feats[n * LED_FN + d] * kw[o * 11 + d];
#pragma unroll
    for (int d = 0; d < 3; ++d)      s += led_pos[n * 3 + d] * kw[o * 11 + 8 + d];
    keys[j] = s;
  }
}

// ---------------------------------------------------------------------------
// Main: TDM stage -> conv1 -> conv2+mean -> queries (WMMA) -> attention tail.
// ---------------------------------------------------------------------------
__global__ __launch_bounds__(256)
void enc_main(const float* __restrict__ rss,       // [B][512][12]
              const float* __restrict__ anchor,    // [B][3]
              const float* __restrict__ led_feats, // [16][8]
              const float* __restrict__ led_pos,   // [16][3]
              const int*   __restrict__ freq_mask, // [12][16]
              const float* __restrict__ c1b,       // [32]
              const float* __restrict__ c2b,       // [64]
              const float* __restrict__ qw_full,   // [64][515]
              const float* __restrict__ qb,        // [64]
              const float* __restrict__ red_w,     // [64][96]
              const float* __restrict__ red_b,     // [64]
              const float* __restrict__ fus_w,     // [64][128]
              const float* __restrict__ fus_b,     // [64]
              const _Float16* __restrict__ w1t,
              const _Float16* __restrict__ w2t,
              const _Float16* __restrict__ qwh,
              const float* __restrict__ keysg,
              float* __restrict__ out)             // [B][64]
{
  extern __shared__ char smem[];
  _Float16* xpad = (_Float16*)(smem + L_XPAD);   // [514][32] row = pos+1
  _Float16* xT   = (_Float16*)(smem + L_XT);     // [16][512]
  _Float16* h1p  = (_Float16*)(smem + L_H1);     // [514][32] row = pos+1
  float* xstage  = (float*)(smem + L_H1);        // [512][12] f32 (aliases h1p)

  __shared__ float keysl[NLED * 64];
  __shared__ float qld[RSS_C * 64];
  __shared__ float tfeat[64];
  __shared__ float attn[RSS_C * NLED];
  __shared__ float sagg[RSS_C * LED_FN];
  __shared__ float comb[128];

  const int tid  = threadIdx.x;
  const int b    = blockIdx.x;
  const int lane = tid & 31;
  const int wv   = tid >> 5;            // wave id 0..7 (wave32)
  const int lr   = lane & 15;
  const int lhi  = lane >> 4;

  // ---- phase 0a: wave 0 kicks the Tensor Data Mover: rss[b] (24576 B) -> LDS
  if (wv == 0) {
    uint64_t ga = (uint64_t)(uintptr_t)(rss + (size_t)b * (W_LEN * RSS_C));
    uint32_t la = (uint32_t)(uintptr_t)xstage;   // flat->LDS: offset = addr[31:0]
    u32x4 g0;
    g0[0] = 1u;                                   // count=1 (valid descriptor)
    g0[1] = la;                                   // lds_addr (bytes)
    g0[2] = (uint32_t)ga;                         // global_addr[31:0]
    g0[3] = (uint32_t)((ga >> 32) & 0x01FFFFFFu)  // global_addr[56:32]
          | 0x80000000u;                          // type=2 ("image")
    i32x8 g1;
    g1[0] = (int)(2u << 16);                      // data_size=2 -> 4-byte elems
    g1[1] = (int)((uint32_t)RSS_C << 16);         // tensor_dim0 = 12
    g1[2] = (int)((uint32_t)W_LEN << 16);         // tensor_dim1[15:0] = 512
    g1[3] = (int)((uint32_t)RSS_C << 16);         // tile_dim0 = 12
    g1[4] = W_LEN;                                // tile_dim1 = 512
    g1[5] = RSS_C;                                // tensor_dim0_stride = 12
    g1[6] = 0;
    g1[7] = 0;
    i32x4 gz4 = {0, 0, 0, 0};                     // 2-D: groups 2/3 unused
    i32x8 gz8 = {0, 0, 0, 0, 0, 0, 0, 0};
    // clang-23 / therock lane: 6-arg form (u32x4, i32x8, i32x4, i32x4, i32x8, i32)
    __builtin_amdgcn_tensor_load_to_lds(g0, g1, gz4, gz4, gz8, 0);
  }

  // ---- phase 0b: init work independent of the DMA target ----
  {
    uint32_t* xw = (uint32_t*)xpad;                 // 514*32 halves = 8224 words
    for (int i = tid; i < 8224; i += 256) xw[i] = 0u;
    uint32_t* tw = (uint32_t*)xT;                   // rows 12..15 zero
    for (int i = tid; i < 1024; i += 256) tw[3072 + i] = 0u;
    if (tid < 64) tfeat[tid] = 0.f;
    for (int i = tid; i < NLED * 64; i += 256) keysl[i] = keysg[i];
  }
  if (wv == 0) __builtin_amdgcn_s_wait_tensorcnt(0);
  __syncthreads();

  // ---- phase 1: convert staged f32 tile into both f16 layouts ----
  for (int i = tid; i < W_LEN * RSS_C; i += 256) {
    int p = i / RSS_C;
    int c = i - p * RSS_C;
    _Float16 h = (_Float16)xstage[i];
    xpad[(p + 1) * 32 + c] = h;        // [pos][ch] for conv1 A-frags
    xT[c * W_LEN + p]      = h;        // [ch][pos] for query A-frags
  }
  __syncthreads();                     // xstage dead beyond this point

  // ---- phase 2: conv1 (12->32, k=3, pad=1) as 3 tap-GEMMs, 64 tile jobs ----
  {                                    // zero h1p halo rows (aliased region now free)
    uint32_t* hw = (uint32_t*)h1p;
    if (tid < 16) { hw[tid] = 0u; hw[513 * 16 + tid] = 0u; }
  }
  for (int job = wv; job < 64; job += 8) {
    const int mt = job >> 1;            // position tile 0..31
    const int nt = job & 1;             // out-channel tile 0..1
    v8f acc = {};
#pragma unroll
    for (int t = 0; t < 3; ++t) {
      const _Float16* ap = xpad + (mt * 16 + lr + t) * 32 + lhi * 8;
      v16h a = load_frag16(ap, ap + 16);
      const _Float16* bp = w1t + (size_t)(t * 32 + nt * 16 + lr) * 32 + lhi * 8;
      v16h bm = load_frag16(bp, bp + 16);
      acc = __builtin_amdgcn_wmma_f32_16x16x32_f16(false, a, false, bm,
                                                   (short)0, acc, false, false);
    }
    const int oc = nt * 16 + lr;
    const float bias = c1b[oc];
#pragma unroll
    for (int m8 = 0; m8 < 8; ++m8) {
      const int m = lhi * 8 + m8;
      float v = fmaxf(acc[m8] + bias, 0.f);
      h1p[(mt * 16 + m + 1) * 32 + oc] = (_Float16)v;   // relu -> f16 stage
    }
  }
  __syncthreads();

  // ---- phase 3: conv2 (32->64, k=3, s=2) + mean, and queries (12x512x64) ----
  for (int job = wv; job < 68; job += 8) {
    if (job < 64) {
      const int mt = job >> 2;          // out-position tile 0..15
      const int nt = job & 3;           // out-channel tile 0..3
      v8f acc = {};
#pragma unroll
      for (int t = 0; t < 3; ++t) {     // input row = 2*p + t (padded)
        const _Float16* ap = h1p + (2 * (mt * 16 + lr) + t) * 32 + lhi * 8;
        v16h a = load_frag16(ap, ap + 16);
        const _Float16* bp = w2t + (size_t)(t * 64 + nt * 16 + lr) * 32 + lhi * 8;
        v16h bm = load_frag16(bp, bp + 16);
        acc = __builtin_amdgcn_wmma_f32_16x16x32_f16(false, a, false, bm,
                                                     (short)0, acc, false, false);
      }
      const int oc = nt * 16 + lr;
      const float bias = c2b[oc];
      float s = 0.f;
#pragma unroll
      for (int m8 = 0; m8 < 8; ++m8) s += fmaxf(acc[m8] + bias, 0.f);
      atomicAdd(&tfeat[oc], s);         // ds_add_f32 into mean accumulator
    } else {
      const int nt = job - 64;          // query N-tile 0..3
      v8f acc = {};
      for (int ks = 0; ks < 16; ++ks) {
        const _Float16* ap = xT + lr * W_LEN + ks * 32 + lhi * 8;
        v16h a = load_frag16(ap, ap + 16);
        const _Float16* bp = qwh + (size_t)(nt * 16 + lr) * 512 + ks * 32 + lhi * 8;
        v16h bm = load_frag16(bp, bp + 16);
        acc = __builtin_amdgcn_wmma_f32_16x16x32_f16(false, a, false, bm,
                                                     (short)0, acc, false, false);
      }
      const int col = nt * 16 + lr;
#pragma unroll
      for (int m8 = 0; m8 < 8; ++m8) {
        const int m = lhi * 8 + m8;
        if (m < RSS_C) qld[m * 64 + col] = acc[m8];
      }
    }
  }
  __syncthreads();

  // ---- phase 4: query bias + anchor columns (512..514) ----
  const float ax = anchor[b * 3 + 0];
  const float ay = anchor[b * 3 + 1];
  const float az = anchor[b * 3 + 2];
  for (int i = tid; i < RSS_C * 64; i += 256) {
    int o = i & 63;
    float add = qb[o] + ax * qw_full[o * 515 + 512]
                      + ay * qw_full[o * 515 + 513]
                      + az * qw_full[o * 515 + 514];
    qld[i] += add;
  }
  __syncthreads();

  // ---- phase 5: scores + dist bias + mask ----
  if (tid < RSS_C * NLED) {
    int r = tid >> 4, n = tid & 15;
    float s = 0.f;
    for (int k = 0; k < 64; ++k) s += qld[r * 64 + k] * keysl[n * 64 + k];
    float d2 = 0.f;
#pragma unroll
    for (int j = 0; j < 3; ++j) {
      float dd = anchor[b * 3 + j] - led_pos[n * 3 + j];
      d2 += dd * dd;
    }
    s += -0.5f * __logf(d2 + 1e-8f);                 // 0.5*log(1/(d2+eps))
    if (freq_mask[r * 16 + n] == 0) s = -3.402823466e38f;
    attn[tid] = s;
  }
  __syncthreads();

  // ---- phase 6: softmax over NLED per (r) ----
  if (tid < RSS_C) {
    float m = -3.402823466e38f;
    for (int n = 0; n < NLED; ++n) m = fmaxf(m, attn[tid * 16 + n]);
    float sum = 0.f;
    for (int n = 0; n < NLED; ++n) {
      float e = __expf(attn[tid * 16 + n] - m);
      attn[tid * 16 + n] = e;
      sum += e;
    }
    float inv = 1.f / sum;
    for (int n = 0; n < NLED; ++n) attn[tid * 16 + n] *= inv;
  }
  __syncthreads();

  // ---- phase 7: s_agg [12][8] and t_feat mean into comb[0..63] ----
  if (tid < RSS_C * LED_FN) {
    int r = tid >> 3, f = tid & 7;
    float s = 0.f;
    for (int n = 0; n < NLED; ++n) s += attn[r * 16 + n] * led_feats[n * LED_FN + f];
    sagg[tid] = s;
  }
  if (tid < 64) comb[tid] = tfeat[tid] * (1.f / 256.f);
  __syncthreads();

  // ---- phase 8: s_feat = relu(sagg @ red_w.T + red_b) -> comb[64..127] ----
  if (tid < 64) {
    float s = red_b[tid];
    for (int i = 0; i < RSS_C * LED_FN; ++i) s += red_w[tid * 96 + i] * sagg[i];
    comb[64 + tid] = fmaxf(s, 0.f);
  }
  __syncthreads();

  // ---- phase 9: fusion -> out[b][0..63] ----
  if (tid < HID) {
    float s = fus_b[tid];
    for (int c = 0; c < 128; ++c) s += fus_w[tid * 128 + c] * comb[c];
    out[(size_t)b * HID + tid] = s;
  }
}

// ---------------------------------------------------------------------------
extern "C" void kernel_launch(void* const* d_in, const int* in_sizes, int n_in,
                              void* d_out, int out_size, void* d_ws, size_t ws_size,
                              hipStream_t stream) {
  const float* rss       = (const float*)d_in[0];
  const float* anchor    = (const float*)d_in[1];
  const float* led_feats = (const float*)d_in[2];
  const float* led_pos   = (const float*)d_in[3];
  const int*   freq_mask = (const int*)d_in[4];
  const float* c1w       = (const float*)d_in[5];
  const float* c1b       = (const float*)d_in[6];
  const float* c2w       = (const float*)d_in[7];
  const float* c2b       = (const float*)d_in[8];
  const float* qw        = (const float*)d_in[9];
  const float* qb        = (const float*)d_in[10];
  const float* kw        = (const float*)d_in[11];
  const float* kb        = (const float*)d_in[12];
  const float* red_w     = (const float*)d_in[13];
  const float* red_b     = (const float*)d_in[14];
  const float* fus_w     = (const float*)d_in[15];
  const float* fus_b     = (const float*)d_in[16];
  float* out = (float*)d_out;

  char* ws = (char*)d_ws;
  _Float16* w1t  = (_Float16*)(ws + WS_W1T);
  _Float16* w2t  = (_Float16*)(ws + WS_W2T);
  _Float16* qwh  = (_Float16*)(ws + WS_QWH);
  float*    keys = (float*)(ws + WS_KEYS);

  enc_setup<<<168, 256, 0, stream>>>(c1w, c2w, qw, led_feats, led_pos, kw, kb,
                                     w1t, w2t, qwh, keys);

  enc_main<<<8192, 256, L_TOTAL, stream>>>(rss, anchor, led_feats, led_pos,
                                           freq_mask, c1b, c2b, qw, qb,
                                           red_w, red_b, fus_w, fus_b,
                                           w1t, w2t, qwh, keys, out);
}
